// MetaMamba_21337397527355
// MI455X (gfx1250) — matrix operand order repaired and checked by reading, not compiled, not run on hardware
//
#include <hip/hip_runtime.h>
#include <math.h>

// ---------------- Problem constants (from reference) ----------------
#define BATCH    256
#define DMODEL   2048
#define DINNER   8192        // 4 * DMODEL
#define DSTATE   16
#define DCONV    4
#define DTRANK   128         // (2048+15)/16
#define XPROJ_N  (DTRANK + 2 * DSTATE)   // 160

typedef __attribute__((ext_vector_type(16))) __bf16 v16bf;
typedef __attribute__((ext_vector_type(8)))  float  v8f;

// ---------------- helpers: f32 -> bf16 tile packing ----------------
__device__ __forceinline__ v16bf pack16(const float* f) {
  v16bf r;
#pragma unroll
  for (int i = 0; i < 16; ++i) r[i] = (__bf16)f[i];
  return r;
}

// B tile fragment: 16 contiguous K-values (column n = lane%16, koff = (lane<16?0:16))
__device__ __forceinline__ v16bf load_b_frag(const float* __restrict__ p) {
  const float4* q = (const float4*)p;
  float4 a = q[0], b = q[1], c = q[2], d = q[3];
  float f[16] = {a.x,a.y,a.z,a.w, b.x,b.y,b.z,b.w,
                 c.x,c.y,c.z,c.w, d.x,d.y,d.z,d.w};
  return pack16(f);
}

// A tile fragment: row m = lane%16, koff=(lane<16?0:16); elems 0..15 = K[koff..koff+15]
__device__ __forceinline__ v16bf load_a_frag(const float* __restrict__ p) {
  const float4* q0 = (const float4*)p;
  float4 a = q0[0], b = q0[1], c = q0[2], d = q0[3];
  float f[16] = {a.x,a.y,a.z,a.w, b.x,b.y,b.z,b.w,
                 c.x,c.y,c.z,c.w, d.x,d.y,d.z,d.w};
  return pack16(f);
}

// ---------------- generic bf16-WMMA GEMM:  C = A(MxK) * B(NxK)^T ----------------
// Wave computes 64(M) x 16(N); block = 8 waves -> 128 N columns.
// grid = dim3(M/64, ceil(N/128)).  EPI: 0 = none, 1 = softplus(c + bias[n])
template <int EPI>
__global__ __launch_bounds__(256) void gemm_bf16_wmma(
    const float* __restrict__ A, int lda,
    const float* __restrict__ Bw, int ldb,
    float* __restrict__ C, int ldc,
    int M, int N, int K,
    const float* __restrict__ bias) {
  const int lane = threadIdx.x & 31;
  const int wave = threadIdx.x >> 5;
  const int tn = blockIdx.y * 8 + wave;       // N tile (16 cols) per wave
  if (tn * 16 >= N) return;                   // wave-uniform exit (EXEC stays full)
  const int n0 = tn * 16;
  const int m0 = blockIdx.x * 64;             // 4 M tiles per wave

  const int r16  = lane & 15;
  const int hi   = lane >> 4;                 // 0 or 1
  const float* Brow = Bw + (size_t)(n0 + r16) * ldb + hi * 16;
  const float* Arow = A  + (size_t)(m0 + r16) * lda + hi * 16;

  v8f acc[4] = {v8f{}, v8f{}, v8f{}, v8f{}};

  for (int k0 = 0; k0 < K; k0 += 32) {
    v16bf bf = load_b_frag(Brow + k0);
#pragma unroll
    for (int t = 0; t < 4; ++t) {
      v16bf af = load_a_frag(Arow + (size_t)t * 16 * lda + k0);
      acc[t] = __builtin_amdgcn_wmma_f32_16x16x32_bf16(
          false, af, false, bf, (short)0, acc[t], false, false);
    }
  }

  const int n     = n0 + r16;
  const int mbase = m0 + (hi << 3);
  float bv = (EPI == 1) ? bias[n] : 0.0f;
#pragma unroll
  for (int t = 0; t < 4; ++t) {
#pragma unroll
    for (int v = 0; v < 8; ++v) {
      float val = acc[t][v];
      if (EPI == 1) {
        val += bv;
        val = (val > 20.0f) ? val : log1pf(__expf(val));   // softplus
      }
      C[(size_t)(mbase + t * 16 + v) * ldc + n] = val;
    }
  }
}

// ---------------- conv-state shift + depthwise conv + SiLU ----------------
// idx = b*DINNER + d.  new_cs = [cs1,cs2,cs3,xi]; xc = silu(dot(new_cs,w_d)+b_d)
__global__ __launch_bounds__(256) void conv_silu_kernel(
    const float* __restrict__ cs_in, const float* __restrict__ xz,
    const float* __restrict__ conv_w, const float* __restrict__ conv_b,
    float* __restrict__ cs_out, float* __restrict__ xc) {
  const int idx = blockIdx.x * 256 + threadIdx.x;   // b*DINNER + d
  const int b = idx >> 13, d = idx & (DINNER - 1);
  const float4 cs = *(const float4*)(cs_in + (size_t)idx * DCONV);
  const float  xi = xz[(size_t)b * (2 * DINNER) + d];
  const float4 w  = *(const float4*)(conv_w + (size_t)d * DCONV);
  float a = cs.y * w.x + cs.z * w.y + cs.w * w.z + xi * w.w + conv_b[d];
  float s = a / (1.0f + __expf(-a));                 // silu
  *(float4*)(cs_out + (size_t)idx * DCONV) = make_float4(cs.y, cs.z, cs.w, xi);
  xc[idx] = s;
}

// ---------------- SSM recurrence + gate ----------------
__global__ __launch_bounds__(256) void ssm_kernel(
    const float* __restrict__ ssm_in, const float* __restrict__ xc,
    const float* __restrict__ dt,     const float* __restrict__ xdb,
    const float* __restrict__ A_log,  const float* __restrict__ D_param,
    const float* __restrict__ xz,
    float* __restrict__ ssm_out, float* __restrict__ y) {
  const int idx = blockIdx.x * 256 + threadIdx.x;   // b*DINNER + d
  const int b = idx >> 13, d = idx & (DINNER - 1);
  const float dtv = dt[idx];
  const float xcv = xc[idx];
  const float* Bp = xdb + (size_t)b * XPROJ_N + DTRANK;
  const float* Cp = Bp + DSTATE;
  const float* Sp = ssm_in + (size_t)idx * DSTATE;
  const float* Ap = A_log + (size_t)d * DSTATE;
  float* So = ssm_out + (size_t)idx * DSTATE;

  float accy = 0.0f;
#pragma unroll
  for (int q = 0; q < 4; ++q) {
    float4 s  = *(const float4*)(Sp + q * 4);
    float4 al = *(const float4*)(Ap + q * 4);
    float4 bm = *(const float4*)(Bp + q * 4);
    float4 cm = *(const float4*)(Cp + q * 4);
    float4 ns;
    ns.x = s.x * __expf(-dtv * __expf(al.x)) + xcv * dtv * bm.x;
    ns.y = s.y * __expf(-dtv * __expf(al.y)) + xcv * dtv * bm.y;
    ns.z = s.z * __expf(-dtv * __expf(al.z)) + xcv * dtv * bm.z;
    ns.w = s.w * __expf(-dtv * __expf(al.w)) + xcv * dtv * bm.w;
    accy += ns.x * cm.x + ns.y * cm.y + ns.z * cm.z + ns.w * cm.w;
    *(float4*)(So + q * 4) = ns;
  }
  accy += D_param[d] * xcv;
  const float z = xz[(size_t)b * (2 * DINNER) + DINNER + d];
  accy *= z / (1.0f + __expf(-z));                   // * silu(z)
  y[idx] = accy;
}

// ---------------- launcher ----------------
extern "C" void kernel_launch(void* const* d_in, const int* in_sizes, int n_in,
                              void* d_out, int out_size, void* d_ws, size_t ws_size,
                              hipStream_t stream) {
  const float* x          = (const float*)d_in[0];
  const float* conv_state = (const float*)d_in[1];
  const float* ssm_state  = (const float*)d_in[2];
  const float* W_in       = (const float*)d_in[3];
  const float* conv_w     = (const float*)d_in[4];
  const float* conv_b     = (const float*)d_in[5];
  const float* W_xproj    = (const float*)d_in[6];
  const float* W_dt       = (const float*)d_in[7];
  const float* b_dt       = (const float*)d_in[8];
  const float* A_log      = (const float*)d_in[9];
  const float* D_param    = (const float*)d_in[10];
  const float* W_out      = (const float*)d_in[11];

  // d_out layout: out | conv_state | ssm_state (flat, return order)
  float* out    = (float*)d_out;                              // 256*2048
  float* cs_out = out + (size_t)BATCH * DMODEL;               // 256*8192*4
  float* ss_out = cs_out + (size_t)BATCH * DINNER * DCONV;    // 256*8192*16

  // workspace
  float* xz  = (float*)d_ws;                                  // 256*16384
  float* xc  = xz  + (size_t)BATCH * 2 * DINNER;              // 256*8192
  float* xdb = xc  + (size_t)BATCH * DINNER;                  // 256*160
  float* dt  = xdb + (size_t)BATCH * XPROJ_N;                 // 256*8192
  float* yv  = dt  + (size_t)BATCH * DINNER;                  // 256*8192

  const int elems = BATCH * DINNER;                           // 2,097,152

  // 1) xz = x @ W_in^T : M=256, N=16384, K=2048
  gemm_bf16_wmma<0><<<dim3(BATCH / 64, (2 * DINNER) / 128), 256, 0, stream>>>(
      x, DMODEL, W_in, DMODEL, xz, 2 * DINNER, BATCH, 2 * DINNER, DMODEL, nullptr);

  // 2) conv shift + silu -> xc, writes new conv_state to output
  conv_silu_kernel<<<elems / 256, 256, 0, stream>>>(
      conv_state, xz, conv_w, conv_b, cs_out, xc);

  // 3) x_db = xc @ W_xproj^T : M=256, N=160, K=8192
  gemm_bf16_wmma<0><<<dim3(BATCH / 64, (XPROJ_N + 127) / 128), 256, 0, stream>>>(
      xc, DINNER, W_xproj, DINNER, xdb, XPROJ_N, BATCH, XPROJ_N, DINNER, nullptr);

  // 4) dt = softplus(x_db[:, :128] @ W_dt^T + b_dt) : M=256, N=8192, K=128
  gemm_bf16_wmma<1><<<dim3(BATCH / 64, DINNER / 128), 256, 0, stream>>>(
      xdb, XPROJ_N, W_dt, DTRANK, dt, DINNER, BATCH, DINNER, DTRANK, b_dt);

  // 5) SSM recurrence + D skip + silu(z) gate -> y; writes new ssm_state
  ssm_kernel<<<elems / 256, 256, 0, stream>>>(
      ssm_state, xc, dt, xdb, A_log, D_param, xz, ss_out, yv);

  // 6) out = y @ W_out^T : M=256, N=2048, K=8192
  gemm_bf16_wmma<0><<<dim3(BATCH / 64, DMODEL / 128), 256, 0, stream>>>(
      yv, DINNER, W_out, DINNER, out, DMODEL, BATCH, DMODEL, DINNER, nullptr);
}
